// CrossCovarianceAttn_37752762532542
// MI455X (gfx1250) — compile-verified, matrix operand order checked by reading, not run on hardware
//
#include <hip/hip_runtime.h>
#include <hip/hip_bf16.h>

typedef __attribute__((ext_vector_type(16))) _Float16 v16h;
typedef __attribute__((ext_vector_type(8)))  _Float16 v8h;
typedef __attribute__((ext_vector_type(8)))  float    v8f;

#define BDIM  8
#define HDIM  8
#define NPOS  4096
#define CDIM  768
#define HD    96

// ---- CDNA5 helpers ----------------------------------------------------------

__device__ inline v8f wmma32(v16h a, v16h b, v8f c) {
  // D = A(16x32 f16) x B(32x16 f16) + C(16x16 f32)
  return __builtin_amdgcn_wmma_f32_16x16x32_f16(
      /*neg_a=*/false, a, /*neg_b=*/false, b,
      /*c_mod=*/(short)0, c, /*reuse_a=*/false, /*reuse_b=*/false);
}

// Async DMA: 16 bytes global -> LDS, tracked by ASYNCcnt (no VGPR staging).
__device__ inline void async_cp16(const _Float16* g, _Float16* l) {
  const uint32_t lds_off = (uint32_t)(uintptr_t)l;   // flat LDS aperture: addr[31:0]
  asm volatile("global_load_async_to_lds_b128 %0, %1, off"
               :: "v"(lds_off), "v"((uint64_t)(uintptr_t)g)
               : "memory");
}

#define WAIT_ASYNC(n) asm volatile("s_wait_asynccnt " #n ::: "memory")

// 16-bit operand fragment from an LDS tile stored [major][k] (k contiguous).
// ISA layout: lanes 0-15 -> major row (lane&15), K halves {0..7,16..23};
// lanes 16-31 -> same rows, K halves {8..15,24..31}.
__device__ inline v16h lds_frag(const _Float16* t, int ld, int majorBase, int lane) {
  const _Float16* p = t + (size_t)(majorBase + (lane & 15)) * ld + ((lane >> 4) << 3);
  v8h lo = *reinterpret_cast<const v8h*>(p);
  v8h hi = *reinterpret_cast<const v8h*>(p + 16);
  v16h f;
#pragma unroll
  for (int j = 0; j < 8; ++j) { f[j] = lo[j]; f[j + 8] = hi[j]; }
  return f;
}

// Fragment where the LDS tile is stored [k][major] (k is the strided dim).
__device__ inline v16h lds_frag_strided(const _Float16* t, int ld, int kBase,
                                        int colBase, int lane) {
  const int col = colBase + (lane & 15);
  const int kb  = kBase + ((lane >> 4) << 3);
  v16h f;
#pragma unroll
  for (int j = 0; j < 8; ++j) {
    f[j]     = t[(size_t)(kb + j) * ld + col];
    f[j + 8] = t[(size_t)(kb + 16 + j) * ld + col];
  }
  return f;
}

// ---- Kernel 1: qkv = x @ w_qkv^T, scattered to q/k/v in (B,H,hd,N) f16 ------

#define GT_M 128
#define GT_N 128
#define GT_K 32
#define GT_LD 40   // 32 + 8 halves pad, keeps 16B alignment

__global__ __launch_bounds__(256) void qkv_kernel(
    const float* __restrict__ x, const float* __restrict__ wqkv,
    _Float16* __restrict__ qb, _Float16* __restrict__ kb,
    _Float16* __restrict__ vb) {
  __shared__ __align__(16) _Float16 As[GT_M * GT_LD];
  __shared__ __align__(16) _Float16 Bs[GT_N * GT_LD];
  const int tid  = threadIdx.x;
  const int lane = tid & 31;
  const int wave = tid >> 5;
  const int m0 = blockIdx.x * GT_M;     // token tile
  const int n0 = blockIdx.y * GT_N;     // out-channel tile (0..2303)
  const int wm = (wave & 3) * 32;       // 4 waves in M
  const int wn = (wave >> 2) * 64;      // 2 waves in N

  const v8f zero = {};
  v8f acc[2][4];
#pragma unroll
  for (int i = 0; i < 2; ++i)
#pragma unroll
    for (int j = 0; j < 4; ++j) acc[i][j] = zero;

  for (int k0 = 0; k0 < CDIM; k0 += GT_K) {
    if (k0 + GT_K < CDIM)
      __builtin_prefetch(&x[(size_t)(m0 + (tid & 127)) * CDIM + k0 + GT_K], 0, 0);
#pragma unroll
    for (int s = 0; s < 4; ++s) {
      const int slot = tid + s * 256;        // 0..1023
      const int r  = slot >> 3;              // tile row 0..127
      const int cc = (slot & 7) * 4;         // k offset
      const float4 a4 = *reinterpret_cast<const float4*>(
          &x[(size_t)(m0 + r) * CDIM + k0 + cc]);
      _Float16* da = &As[r * GT_LD + cc];
      da[0] = (_Float16)a4.x; da[1] = (_Float16)a4.y;
      da[2] = (_Float16)a4.z; da[3] = (_Float16)a4.w;
      const float4 b4 = *reinterpret_cast<const float4*>(
          &wqkv[(size_t)(n0 + r) * CDIM + k0 + cc]);
      _Float16* db = &Bs[r * GT_LD + cc];
      db[0] = (_Float16)b4.x; db[1] = (_Float16)b4.y;
      db[2] = (_Float16)b4.z; db[3] = (_Float16)b4.w;
    }
    __syncthreads();
    v16h af[2], bf[4];
#pragma unroll
    for (int i = 0; i < 2; ++i) af[i] = lds_frag(As, GT_LD, wm + i * 16, lane);
#pragma unroll
    for (int j = 0; j < 4; ++j) bf[j] = lds_frag(Bs, GT_LD, wn + j * 16, lane);
#pragma unroll
    for (int i = 0; i < 2; ++i)
#pragma unroll
      for (int j = 0; j < 4; ++j) acc[i][j] = wmma32(af[i], bf[j], acc[i][j]);
    __syncthreads();
  }

  // C layout: VGPR r -> M=r (lanes 0-15) / M=r+8 (lanes 16-31); N = lane&15.
  const int e   = lane & 15;
  const int hi8 = (lane >> 4) << 3;
#pragma unroll
  for (int i = 0; i < 2; ++i) {
    const int tok0 = m0 + wm + i * 16 + hi8;     // 8 consecutive tokens
    const int bidx = tok0 >> 12;                 // /4096
    const int n    = tok0 & (NPOS - 1);
#pragma unroll
    for (int j = 0; j < 4; ++j) {
      const int c  = n0 + wn + j * 16 + e;       // 0..2303
      const int s  = c / CDIM;                   // 0=q 1=k 2=v
      const int ch = c - s * CDIM;
      const int h  = ch / HD;
      const int d  = ch - h * HD;
      _Float16* base = (s == 0) ? qb : ((s == 1) ? kb : vb);
      _Float16* p = base + (((size_t)(bidx * HDIM + h) * HD + d) * NPOS + n);
      v8h o;
#pragma unroll
      for (int r = 0; r < 8; ++r) o[r] = (_Float16)acc[i][j][r];
      *reinterpret_cast<v8h*>(p) = o;
    }
  }
}

// ---- Kernel 2: inverse L2 norms over the token dim --------------------------

__global__ __launch_bounds__(256) void norm_kernel(
    const _Float16* __restrict__ qb, const _Float16* __restrict__ kb,
    float* __restrict__ invq, float* __restrict__ invk) {
  __shared__ float red[256];
  const int row = blockIdx.x;                 // 0..6143  (= bh*96 + d)
  const _Float16* src = (blockIdx.y == 0) ? qb : kb;
  const _Float16* p = src + (size_t)row * NPOS;
  float s = 0.f;
  for (int i = threadIdx.x; i < NPOS; i += 256) {
    const float v = (float)p[i];
    s += v * v;
  }
  red[threadIdx.x] = s;
  __syncthreads();
  for (int st = 128; st > 0; st >>= 1) {
    if (threadIdx.x < st) red[threadIdx.x] += red[threadIdx.x + st];
    __syncthreads();
  }
  if (threadIdx.x == 0) {
    const float nrm = fmaxf(sqrtf(red[0]), 1e-12f);
    ((blockIdx.y == 0) ? invq : invk)[row] = 1.f / nrm;
  }
}

// ---- Kernel 3: per-head Gram matrix + softmax -> P (f16) --------------------
// One WG per (b,h). S = q k^T over K=4096 streamed through double-buffered
// async Global->LDS copies (ASYNCcnt), scaled by invq*invk*temp, softmax.

#define GK_CHUNK 64
#define GK_LD    72           // 64 + 8 halves pad
#define GK_BUFB  (HD * GK_LD * 2)   // bytes per 96x72 f16 buffer = 13824

__global__ __launch_bounds__(256) void gram_kernel(
    const _Float16* __restrict__ qb, const _Float16* __restrict__ kb,
    const float* __restrict__ invq, const float* __restrict__ invk,
    const float* __restrict__ temperature, _Float16* __restrict__ pbuf) {
  __shared__ __align__(16) char smem[4 * GK_BUFB];   // 55296 B
  _Float16* qs0 = reinterpret_cast<_Float16*>(smem);
  _Float16* ks0 = reinterpret_cast<_Float16*>(smem + GK_BUFB);
  _Float16* qs1 = reinterpret_cast<_Float16*>(smem + 2 * GK_BUFB);
  _Float16* ks1 = reinterpret_cast<_Float16*>(smem + 3 * GK_BUFB);
  float* Sm = reinterpret_cast<float*>(smem);        // reused after K loop

  const int tid  = threadIdx.x;
  const int lane = tid & 31;
  const int wave = tid >> 5;
  const int bh = blockIdx.x;                 // 0..63
  const int h  = bh & (HDIM - 1);
  const _Float16* qg = qb + (size_t)bh * HD * NPOS;
  const _Float16* kg = kb + (size_t)bh * HD * NPOS;

  // 96 rows x 64 halves per operand = 768 b128 transfers / 256 thr = 3 each,
  // so 6 async instructions per wave per chunk.
  auto issue_chunk = [&](int nc, int buf) {
    _Float16* qs = buf ? qs1 : qs0;
    _Float16* ks = buf ? ks1 : ks0;
#pragma unroll
    for (int s = 0; s < 3; ++s) {
      const int slot = tid + s * 256;        // 0..767
      const int r  = slot >> 3;              // row 0..95
      const int c8 = (slot & 7) * 8;         // half offset 0..56
      async_cp16(&qg[(size_t)r * NPOS + nc + c8], &qs[r * GK_LD + c8]);
      async_cp16(&kg[(size_t)r * NPOS + nc + c8], &ks[r * GK_LD + c8]);
    }
  };

  const v8f zero = {};
  v8f acc[5];
#pragma unroll
  for (int u = 0; u < 5; ++u) acc[u] = zero;

  issue_chunk(0, 0);
  const int NCHUNK = NPOS / GK_CHUNK;        // 64
  for (int c = 0; c < NCHUNK; ++c) {
    const int cur = c & 1;
    if (c + 1 < NCHUNK) {
      issue_chunk((c + 1) * GK_CHUNK, 1 - cur);
      WAIT_ASYNC(6);                         // chunk c landed; next 6 in flight
    } else {
      WAIT_ASYNC(0);
    }
    __syncthreads();
    const _Float16* qs = cur ? qs1 : qs0;
    const _Float16* ks = cur ? ks1 : ks0;
#pragma unroll
    for (int kk = 0; kk < GK_CHUNK; kk += 32) {
#pragma unroll
      for (int u = 0; u < 5; ++u) {
        const int t = wave + u * 8;
        if (t < 36) {
          const int dt = (t / 6) * 16;
          const int et = (t % 6) * 16;
          v16h a = lds_frag(qs + kk, GK_LD, dt, lane);
          v16h b = lds_frag(ks + kk, GK_LD, et, lane);
          acc[u] = wmma32(a, b, acc[u]);
        }
      }
    }
    __syncthreads();
  }

  // scale + deposit into Sm (96 x 100 f32, reusing the staging LDS)
  const float temp = temperature[h];
  const int e0  = lane & 15;
  const int hi8 = (lane >> 4) << 3;
#pragma unroll
  for (int u = 0; u < 5; ++u) {
    const int t = wave + u * 8;
    if (t < 36) {
      const int dt = (t / 6) * 16;
      const int et = (t % 6) * 16;
      const int e  = et + e0;
      const float ike = invk[bh * HD + e] * temp;
#pragma unroll
      for (int r = 0; r < 8; ++r) {
        const int d = dt + hi8 + r;
        Sm[d * 100 + e] = acc[u][r] * invq[bh * HD + d] * ike;
      }
    }
  }
  __syncthreads();

  // softmax rows of length 96, write P as f16
  _Float16* pb = pbuf + (size_t)bh * HD * HD;
  if (tid < HD) {
    float mx = -3.4e38f;
    for (int e = 0; e < HD; ++e) mx = fmaxf(mx, Sm[tid * 100 + e]);
    float sum = 0.f;
    for (int e = 0; e < HD; ++e) {
      const float ex = __expf(Sm[tid * 100 + e] - mx);
      Sm[tid * 100 + e] = ex;
      sum += ex;
    }
    const float inv = 1.f / sum;
    for (int e = 0; e < HD; ++e)
      pb[tid * HD + e] = (_Float16)(Sm[tid * 100 + e] * inv);
  }
}

// ---- Kernel 4: out = P @ v, per (head, 128-token chunk) ---------------------

#define AV_LD 136

__global__ __launch_bounds__(256) void av_kernel(
    const _Float16* __restrict__ pbuf, const _Float16* __restrict__ vb,
    _Float16* __restrict__ obuf) {
  __shared__ __align__(16) _Float16 Ps[HD * HD];      // 18432 B
  __shared__ __align__(16) _Float16 vs[HD * AV_LD];   // 26112 B
  const int tid  = threadIdx.x;
  const int lane = tid & 31;
  const int wave = tid >> 5;
  const int nc0 = blockIdx.x * 128;          // token chunk
  const int bh  = blockIdx.y;                // 0..63
  const int bidx = bh >> 3;
  const int h    = bh & (HDIM - 1);
  const _Float16* pg = pbuf + (size_t)bh * HD * HD;
  const _Float16* vg = vb + (size_t)bh * HD * NPOS;

  // async DMA both operand tiles into LDS
#pragma unroll
  for (int s = 0; s < 5; ++s) {
    const int slot = tid + s * 256;          // 0..1151 valid
    if (slot < 1152) {
      const int r  = slot / 12;
      const int c8 = (slot % 12) * 8;
      async_cp16(&pg[(size_t)r * HD + c8], &Ps[r * HD + c8]);
    }
  }
#pragma unroll
  for (int s = 0; s < 6; ++s) {
    const int slot = tid + s * 256;          // 0..1535
    const int r  = slot >> 4;
    const int c8 = (slot & 15) * 8;
    async_cp16(&vg[(size_t)r * NPOS + nc0 + c8], &vs[r * AV_LD + c8]);
  }
  WAIT_ASYNC(0);
  __syncthreads();

  const v8f zero = {};
  v8f acc[6];
#pragma unroll
  for (int u = 0; u < 6; ++u) acc[u] = zero;

#pragma unroll
  for (int kk = 0; kk < HD; kk += 32) {
#pragma unroll
    for (int u = 0; u < 6; ++u) {
      const int t  = wave + u * 8;           // 0..47
      const int dt = (t >> 3) * 16;
      const int nt = (t & 7) * 16;
      v16h a = lds_frag(Ps + kk, HD, dt, lane);
      v16h b = lds_frag_strided(vs, AV_LD, kk, nt, lane);
      acc[u] = wmma32(a, b, acc[u]);
    }
  }

  const int e0  = lane & 15;
  const int hi8 = (lane >> 4) << 3;
#pragma unroll
  for (int u = 0; u < 6; ++u) {
    const int t  = wave + u * 8;
    const int dt = (t >> 3) * 16;
    const int nt = (t & 7) * 16;
    const int n  = nc0 + nt + e0;
    const int d0 = dt + hi8;                 // 8 consecutive channels
    _Float16* p = obuf + ((size_t)(bidx * NPOS + n) * CDIM + h * HD + d0);
    v8h o;
#pragma unroll
    for (int r = 0; r < 8; ++r) o[r] = (_Float16)acc[u][r];
    *reinterpret_cast<v8h*>(p) = o;
  }
}

// ---- Kernel 5: final projection out @ w_proj^T + b_proj (f32 out) -----------
// A tile (already f16) arrives via async DMA, overlapped with the f32->f16
// conversion of the B tile on the VALU.

__global__ __launch_bounds__(256) void proj_kernel(
    const _Float16* __restrict__ ag, const float* __restrict__ wproj,
    const float* __restrict__ bproj, float* __restrict__ out) {
  __shared__ __align__(16) _Float16 As[GT_M * GT_LD];
  __shared__ __align__(16) _Float16 Bs[GT_N * GT_LD];
  const int tid  = threadIdx.x;
  const int lane = tid & 31;
  const int wave = tid >> 5;
  const int m0 = blockIdx.x * GT_M;
  const int n0 = blockIdx.y * GT_N;          // 0..767
  const int wm = (wave & 3) * 32;
  const int wn = (wave >> 2) * 64;

  const v8f zero = {};
  v8f acc[2][4];
#pragma unroll
  for (int i = 0; i < 2; ++i)
#pragma unroll
    for (int j = 0; j < 4; ++j) acc[i][j] = zero;

  for (int k0 = 0; k0 < CDIM; k0 += GT_K) {
#pragma unroll
    for (int s = 0; s < 2; ++s) {            // A: async f16 DMA to LDS
      const int slot = tid + s * 256;        // 0..511
      const int r  = slot >> 2;
      const int c8 = (slot & 3) * 8;
      async_cp16(&ag[(size_t)(m0 + r) * CDIM + k0 + c8], &As[r * GT_LD + c8]);
    }
#pragma unroll
    for (int s = 0; s < 4; ++s) {            // B: f32 -> f16 on VALU (overlaps DMA)
      const int slot = tid + s * 256;
      const int r  = slot >> 3;
      const int cc = (slot & 7) * 4;
      const float4 b4 = *reinterpret_cast<const float4*>(
          &wproj[(size_t)(n0 + r) * CDIM + k0 + cc]);
      _Float16* db = &Bs[r * GT_LD + cc];
      db[0] = (_Float16)b4.x; db[1] = (_Float16)b4.y;
      db[2] = (_Float16)b4.z; db[3] = (_Float16)b4.w;
    }
    WAIT_ASYNC(0);
    __syncthreads();
    v16h af[2], bf[4];
#pragma unroll
    for (int i = 0; i < 2; ++i) af[i] = lds_frag(As, GT_LD, wm + i * 16, lane);
#pragma unroll
    for (int j = 0; j < 4; ++j) bf[j] = lds_frag(Bs, GT_LD, wn + j * 16, lane);
#pragma unroll
    for (int i = 0; i < 2; ++i)
#pragma unroll
      for (int j = 0; j < 4; ++j) acc[i][j] = wmma32(af[i], bf[j], acc[i][j]);
    __syncthreads();
  }

  const int e   = lane & 15;
  const int hi8 = (lane >> 4) << 3;
#pragma unroll
  for (int i = 0; i < 2; ++i) {
    const int tok0 = m0 + wm + i * 16 + hi8;
#pragma unroll
    for (int j = 0; j < 4; ++j) {
      const int c = n0 + wn + j * 16 + e;
      const float bc = bproj[c];
#pragma unroll
      for (int r = 0; r < 8; ++r)
        out[(size_t)(tok0 + r) * CDIM + c] = acc[i][j][r] + bc;
    }
  }
}

// ---- Host launch ------------------------------------------------------------

extern "C" void kernel_launch(void* const* d_in, const int* in_sizes, int n_in,
                              void* d_out, int out_size, void* d_ws, size_t ws_size,
                              hipStream_t stream) {
  (void)in_sizes; (void)n_in; (void)out_size; (void)ws_size;
  const float* x     = (const float*)d_in[0];
  const float* wqkv  = (const float*)d_in[1];
  const float* temp  = (const float*)d_in[2];
  const float* wproj = (const float*)d_in[3];
  const float* bproj = (const float*)d_in[4];
  float* out = (float*)d_out;

  char* w = (char*)d_ws;
  size_t off = 0;
  auto take = [&](size_t bytes) -> char* {
    char* p = w + off;
    off = (off + bytes + 255) & ~size_t(255);
    return p;
  };
  const size_t QKV_ELEMS = (size_t)BDIM * HDIM * HD * NPOS;   // 25,165,824
  _Float16* qb   = (_Float16*)take(QKV_ELEMS * 2);
  _Float16* kb   = (_Float16*)take(QKV_ELEMS * 2);
  _Float16* vb   = (_Float16*)take(QKV_ELEMS * 2);
  float*    invq = (float*)take((size_t)BDIM * HDIM * HD * 4);
  float*    invk = (float*)take((size_t)BDIM * HDIM * HD * 4);
  _Float16* pbuf = (_Float16*)take((size_t)BDIM * HDIM * HD * HD * 2);
  _Float16* obuf = (_Float16*)take(QKV_ELEMS * 2);

  qkv_kernel<<<dim3((BDIM * NPOS) / GT_M, (3 * CDIM) / GT_N), 256, 0, stream>>>(
      x, wqkv, qb, kb, vb);
  norm_kernel<<<dim3(BDIM * HDIM * HD, 2), 256, 0, stream>>>(qb, kb, invq, invk);
  gram_kernel<<<dim3(BDIM * HDIM), 256, 0, stream>>>(qb, kb, invq, invk, temp, pbuf);
  av_kernel<<<dim3(NPOS / 128, BDIM * HDIM), 256, 0, stream>>>(pbuf, vb, obuf);
  proj_kernel<<<dim3((BDIM * NPOS) / GT_M, CDIM / GT_N), 256, 0, stream>>>(
      obuf, wproj, bproj, out);
}
